// ResNet_cifar10_83150566850848
// MI455X (gfx1250) — 8-bit matrix operand order repaired and checked by reading, compiled, not run on hardware
//
#include <hip/hip_runtime.h>
#include <cstdint>
#include <cstddef>


typedef __attribute__((ext_vector_type(8))) int v8i;

#define BN_EPS 1e-5f

__global__ void pack_wsign(const float* __restrict__ w, signed char* __restrict__ wpk,
                           int K, int Kp, int O) {
  int ntiles = O >> 4;
  long total = (long)Kp * O;
  long idx = (long)blockIdx.x * blockDim.x + threadIdx.x;
  if (idx >= total) return;
  int j = (int)(idx & 31);
  long t = idx >> 5;
  int lane = (int)(t & 31);
  t >>= 5;
  int nt = (int)(t % ntiles);
  int kc = (int)(t / ntiles);
  int ka = kc * 64 + ((lane >> 4) << 3) + (j & 7) + ((j >> 3) << 4);
  int o  = nt * 16 + (lane & 15);
  signed char v = 0;
  if (ka < K) v = (w[(long)o * K + ka] >= 0.f) ? 1 : -1;
  wpk[idx] = v;
}

template<int KH>
__global__ __launch_bounds__(160)
void binconv_wmma(const signed char* __restrict__ sin,
                  const signed char* __restrict__ wpk,
                  float* __restrict__ out,
                  int Hp, int Wp, int Cin, int Ho, int Wo, int O,
                  int stride, int off, int nch,
                  unsigned* __restrict__ regc, int regen) {
  __shared__ __align__(16) signed char smA[16 * 64];
  __shared__ int rowN[16], rowY[16], rowX[16];
  __shared__ unsigned sred[160];

  const int tid  = threadIdx.x;
  const int lane = tid & 31;
  const int wave = tid >> 5;
  const int m0   = blockIdx.x * 16;
  const int ntile = blockIdx.y * 5 + wave;
  const int n0   = ntile * 16;
  const int ntiles = O >> 4;
  const int HoWo = Ho * Wo;

  if (tid < 16) {
    int m = m0 + tid;
    int n = m / HoWo;
    int r = m - n * HoWo;
    int y = r / Wo;
    rowN[tid] = n; rowY[tid] = y; rowX[tid] = r - y * Wo;
  }
  __syncthreads();

  v8i acc;
  #pragma unroll
  for (int j = 0; j < 8; ++j) acc[j] = 0;
  unsigned odd = 0;

  const int rr = lane & 15;
  const int hh = lane >> 4;

  for (int kc = 0; kc < nch; ++kc) {
    __syncthreads();
    for (int i = tid; i < 1024; i += 160) {
      int r = i >> 6, k = i & 63;
      int ka = kc * 64 + k;
      int c, dy, dx;
      if (KH == 1) { c = ka; dy = 0; dx = 0; }
      else { c = ka / 9; int r9 = ka - c * 9; dy = r9 / 3; dx = r9 - dy * 3; }
      signed char v = 0;
      if (c < Cin) {
        int iy = rowY[r] * stride + dy + off;
        int ix = rowX[r] * stride + dx + off;
        v = sin[(((long)rowN[r] * Hp + iy) * Wp + ix) * Cin + c];
      }
      smA[i] = v;
    }
    __syncthreads();

    const int2* ap = reinterpret_cast<const int2*>(smA + rr * 64 + hh * 8);
    int2 t0 = ap[0], t1 = ap[2], t2 = ap[4], t3 = ap[6];
    v8i A;
    A[0] = t0.x; A[1] = t0.y; A[2] = t1.x; A[3] = t1.y;
    A[4] = t2.x; A[5] = t2.y; A[6] = t3.x; A[7] = t3.y;

    const int4* bp = reinterpret_cast<const int4*>(
        wpk + (((long)(kc * ntiles + ntile) * 32 + lane) << 5));
    int4 b0 = bp[0], b1 = bp[1];
    v8i B;
    B[0] = b0.x; B[1] = b0.y; B[2] = b0.z; B[3] = b0.w;
    B[4] = b1.x; B[5] = b1.y; B[6] = b1.z; B[7] = b1.w;

    v8i z;
    #pragma unroll
    for (int j = 0; j < 8; ++j) z[j] = 0;

    v8i ps = __builtin_amdgcn_wmma_i32_16x16x64_iu8(
         true, A,  true, B, z,
         false,  false);

    #pragma unroll
    for (int j = 0; j < 8; ++j) {
      int p = ps[j];
      int o = p & 1;
      int corr = o ? ((p & 2) ? 1 : -1) : 0;
      acc[j] += p + corr;
      odd += (unsigned)o;
    }
  }

  #pragma unroll
  for (int j = 0; j < 8; ++j) {
    int a = acc[j];
    a = a < -128 ? -128 : (a > 127 ? 127 : a);
    int ml = hh * 8 + j;
    out[(long)(m0 + ml) * O + n0 + rr] = (float)a;
  }

  if (regen) {
    sred[tid] = odd;
    __syncthreads();
    if (tid == 0) {
      unsigned s = 0;
      for (int i = 0; i < 160; ++i) s += sred[i];
      atomicAdd(regc, s);
    }
  }
}

__global__ void conv1_fp(const float* __restrict__ x, const float* __restrict__ w,
                         float* __restrict__ out) {
  long idx = (long)blockIdx.x * blockDim.x + threadIdx.x;
  if (idx >= (long)64 * 32 * 32 * 80) return;
  int o = (int)(idx % 80);
  long m = idx / 80;
  int xx = (int)(m & 31), yy = (int)((m >> 5) & 31), n = (int)(m >> 10);
  float acc = 0.f;
  for (int c = 0; c < 3; ++c)
    for (int dy = 0; dy < 3; ++dy) {
      int iy = yy + dy - 1;
      if ((unsigned)iy >= 32u) continue;
      for (int dx = 0; dx < 3; ++dx) {
        int ix = xx + dx - 1;
        if ((unsigned)ix >= 32u) continue;
        acc += x[((n * 3 + c) * 32 + iy) * 32 + ix] *
               w[((o * 3 + c) * 3 + dy) * 3 + dx];
      }
    }
  out[m * 80 + o] = acc;
}

__global__ __launch_bounds__(256)
void bn_stats(const float* __restrict__ in, int M, int C,
              float* __restrict__ mean, float* __restrict__ inv) {
  __shared__ float s1[256], s2[256];
  int c = blockIdx.x, t = threadIdx.x;
  float a = 0.f, b = 0.f;
  for (int m = t; m < M; m += 256) {
    float v = in[(long)m * C + c];
    a += v; b += v * v;
  }
  s1[t] = a; s2[t] = b;
  __syncthreads();
  for (int s = 128; s > 0; s >>= 1) {
    if (t < s) { s1[t] += s1[t + s]; s2[t] += s2[t + s]; }
    __syncthreads();
  }
  if (t == 0) {
    float mu = s1[0] / (float)M;
    float var = s2[0] / (float)M - mu * mu;
    mean[c] = mu;
    inv[c] = rsqrtf(var + BN_EPS);
  }
}

__global__ void bn_norm(const float* __restrict__ in, const float* __restrict__ mean,
                        const float* __restrict__ inv, const float* __restrict__ g,
                        const float* __restrict__ b, float* __restrict__ fpout,
                        signed char* __restrict__ sgnout,
                        long M, int C, int doclip, int H, int W, int Hp, int Wp, int P) {
  long idx = (long)blockIdx.x * blockDim.x + threadIdx.x;
  if (idx >= M * C) return;
  int c = (int)(idx % C);
  long m = idx / C;
  float v = (in[idx] - mean[c]) * inv[c] * g[c] + b[c];
  if (doclip) v = fminf(1.f, fmaxf(-1.f, v));
  if (fpout) fpout[idx] = v;
  if (sgnout) {
    int hw = H * W;
    int n = (int)(m / hw);
    int r = (int)(m - (long)n * hw);
    int y = r / W, xx = r - y * W;
    sgnout[(((long)n * Hp + y + P) * Wp + xx + P) * C + c] = (v >= 0.f) ? 1 : -1;
  }
}

__global__ void add_k(const float* __restrict__ a, const float* __restrict__ b,
                      float* __restrict__ c, long n) {
  long i = (long)blockIdx.x * blockDim.x + threadIdx.x;
  if (i < n) c[i] = a[i] + b[i];
}

__global__ void pool_k(const float* __restrict__ in, float* __restrict__ out) {
  int idx = blockIdx.x * blockDim.x + threadIdx.x;
  if (idx >= 64 * 320) return;
  int c = idx % 320, n = idx / 320;
  float s = 0.f;
  for (int p = 0; p < 64; ++p) s += in[((long)n * 64 + p) * 320 + c];
  out[idx] = s * (1.f / 64.f);
}

__global__ void bn1d_feat(const float* __restrict__ in, const float* __restrict__ g,
                          const float* __restrict__ b, float* __restrict__ out,
                          int R, int C, int doclip) {
  int c = blockIdx.x * blockDim.x + threadIdx.x;
  if (c >= C) return;
  float s = 0.f, q = 0.f;
  for (int r = 0; r < R; ++r) { float v = in[(long)r * C + c]; s += v; q += v * v; }
  float mu = s / (float)R;
  float var = q / (float)R - mu * mu;
  float iv = rsqrtf(var + BN_EPS);
  for (int r = 0; r < R; ++r) {
    float v = (in[(long)r * C + c] - mu) * iv * g[c] + b[c];
    if (doclip) v = fminf(1.f, fmaxf(-1.f, v));
    out[(long)r * C + c] = v;
  }
}

__global__ void fc_k(const float* __restrict__ in, const float* __restrict__ w,
                     const float* __restrict__ bias, float* __restrict__ out) {
  int idx = blockIdx.x * blockDim.x + threadIdx.x;
  if (idx >= 640) return;
  int o = idx % 10, n = idx / 10;
  float a = bias[o];
  for (int k = 0; k < 320; ++k) a += in[n * 320 + k] * w[o * 320 + k];
  out[idx] = a;
}

__global__ void lsm_k(const float* __restrict__ in, float* __restrict__ out,
                      const unsigned* __restrict__ regc) {
  int n = blockIdx.x * blockDim.x + threadIdx.x;
  if (n == 0) out[640] = (float)(*regc);
  if (n >= 64) return;
  float mx = -1e30f;
  for (int o = 0; o < 10; ++o) mx = fmaxf(mx, in[n * 10 + o]);
  float s = 0.f;
  for (int o = 0; o < 10; ++o) s += __expf(in[n * 10 + o] - mx);
  float l = __logf(s);
  for (int o = 0; o < 10; ++o) out[n * 10 + o] = in[n * 10 + o] - mx - l;
}

extern "C" void kernel_launch(void* const* d_in, const int* in_sizes, int n_in,
                              void* d_out, int out_size, void* d_ws, size_t ws_size,
                              hipStream_t stream) {
  (void)in_sizes; (void)out_size;
  if (n_in < 52) return;

  char* ws = (char*)d_ws;
  size_t off = 0;
  auto alloc = [&](size_t bytes) -> char* {
    char* p = ws + off;
    off = (off + bytes + 255) & ~(size_t)255;
    return p;
  };

  const size_t ACT = (size_t)65536 * 80 * 4;
  const size_t SGN = (size_t)64 * 34 * 34 * 80;
  float* actX = (float*)alloc(ACT);
  float* actT = (float*)alloc(ACT);
  float* actR = (float*)alloc(ACT);
  signed char* sgnA = (signed char*)alloc(SGN);
  signed char* sgnB = (signed char*)alloc(SGN);
  float* mean   = (float*)alloc(512 * 4);
  float* invs   = (float*)alloc(512 * 4);
  float* pooled  = (float*)alloc(64 * 320 * 4);
  float* pooled2 = (float*)alloc(64 * 320 * 4);
  float* fcout   = (float*)alloc(64 * 10 * 4);
  float* bnout   = (float*)alloc(64 * 10 * 4);
  unsigned* regc = (unsigned*)alloc(256);
  signed char* wpk_base = (signed char*)alloc((size_t)8 << 20);
  if (off > ws_size) return;

  const float* x = (const float*)d_in[0];
  float* out = (float*)d_out;

  signed char* wseg = wpk_base;
  auto packw = [&](const float* w, int K, int O) -> signed char* {
    int Kp = ((K + 63) >> 6) << 6;
    long total = (long)Kp * O;
    signed char* dst = wseg;
    wseg += total;
    pack_wsign<<<(unsigned)((total + 255) / 256), 256, 0, stream>>>(w, dst, K, Kp, O);
    return dst;
  };
  auto binconv = [&](const signed char* sin, const signed char* wpk, float* o,
                     int Hp, int Cin, int Ho, int O, int kh, int stride, int offp,
                     int K, int regen) {
    int nch = (K + 63) >> 6;
    dim3 g((unsigned)((64 * Ho * Ho) / 16), (unsigned)((O >> 4) / 5));
    if (kh == 3)
      binconv_wmma<3><<<g, 160, 0, stream>>>(sin, wpk, o, Hp, Hp, Cin, Ho, Ho, O,
                                             stride, offp, nch, regc, regen);
    else
      binconv_wmma<1><<<g, 160, 0, stream>>>(sin, wpk, o, Hp, Hp, Cin, Ho, Ho, O,
                                             stride, offp, nch, regc, regen);
  };
  auto norm = [&](const float* in, const float* g, const float* b, float* fpo,
                  signed char* sg, long M, int C, int clip, int H, int Hp) {
    bn_norm<<<(unsigned)((M * C + 255) / 256), 256, 0, stream>>>(
        in, mean, invs, g, b, fpo, sg, M, C, clip, H, H, Hp, Hp, 1);
  };

  hipMemsetAsync(regc, 0, 4, stream);

  conv1_fp<<<(64 * 32 * 32 * 80 + 255) / 256, 256, 0, stream>>>(
      x, (const float*)d_in[1], actT);
  bn_stats<<<80, 256, 0, stream>>>(actT, 65536, 80, mean, invs);
  hipMemsetAsync(sgnA, 0, (size_t)64 * 34 * 34 * 80, stream);
  norm(actT, (const float*)d_in[2], (const float*)d_in[3], actX, sgnA,
       65536, 80, 1, 32, 34);

  const int baseIdx[6] = {4, 10, 16, 25, 31, 40};
  const int CIN[6] = {80, 80, 80, 160, 160, 320};
  const int COUT[6] = {80, 80, 160, 160, 320, 320};
  const int STR[6] = {1, 1, 2, 1, 2, 1};
  const int HASDS[6] = {0, 0, 1, 0, 1, 0};
  const int DBT[6] = {1, 1, 1, 1, 1, 0};
  const int HIN[6] = {32, 32, 32, 16, 16, 8};

  for (int i = 0; i < 6; ++i) {
    int bi = baseIdx[i];
    int Hi = HIN[i], Ho = Hi / STR[i];
    int Hpi = Hi + 2, Hpo = Ho + 2;
    long Mout = (long)64 * Ho * Ho;
    int K1 = CIN[i] * 9, K2 = COUT[i] * 9;

    signed char* w1 = packw((const float*)d_in[bi + 0], K1, COUT[i]);
    binconv(sgnA, w1, actT, Hpi, CIN[i], Ho, COUT[i], 3, STR[i], 0, K1, 1);
    bn_stats<<<COUT[i], 256, 0, stream>>>(actT, (int)Mout, COUT[i], mean, invs);
    hipMemsetAsync(sgnB, 0, (size_t)64 * Hpo * Hpo * COUT[i], stream);
    norm(actT, (const float*)d_in[bi + 1], (const float*)d_in[bi + 2],
         nullptr, sgnB, Mout, COUT[i], 1, Ho, Hpo);

    signed char* w2 = packw((const float*)d_in[bi + 3], K2, COUT[i]);
    binconv(sgnB, w2, actR, Hpo, COUT[i], Ho, COUT[i], 3, 1, 0, K2, HASDS[i] ? 0 : 1);

    if (HASDS[i]) {
      int Kd = CIN[i];
      signed char* wd = packw((const float*)d_in[bi + 6], Kd, COUT[i]);
      binconv(sgnA, wd, actT, Hpi, CIN[i], Ho, COUT[i], 1, STR[i], 1, Kd, 1);
      bn_stats<<<COUT[i], 256, 0, stream>>>(actT, (int)Mout, COUT[i], mean, invs);
      norm(actT, (const float*)d_in[bi + 7], (const float*)d_in[bi + 8],
           actT, nullptr, Mout, COUT[i], 0, Ho, Hpo);
      add_k<<<(unsigned)((Mout * COUT[i] + 255) / 256), 256, 0, stream>>>(
          actR, actT, actT, Mout * COUT[i]);
    } else {
      add_k<<<(unsigned)((Mout * COUT[i] + 255) / 256), 256, 0, stream>>>(
          actR, actX, actT, Mout * COUT[i]);
    }

    if (DBT[i]) {
      bn_stats<<<COUT[i], 256, 0, stream>>>(actT, (int)Mout, COUT[i], mean, invs);
      hipMemsetAsync(sgnA, 0, (size_t)64 * Hpo * Hpo * COUT[i], stream);
      norm(actT, (const float*)d_in[bi + 4], (const float*)d_in[bi + 5],
           actX, sgnA, Mout, COUT[i], 1, Ho, Hpo);
    }
  }

  pool_k<<<80, 256, 0, stream>>>(actT, pooled);
  bn1d_feat<<<5, 64, 0, stream>>>(pooled, (const float*)d_in[46],
                                  (const float*)d_in[47], pooled2, 64, 320, 1);
  fc_k<<<3, 256, 0, stream>>>(pooled2, (const float*)d_in[50],
                              (const float*)d_in[51], fcout);
  bn1d_feat<<<1, 64, 0, stream>>>(fcout, (const float*)d_in[48],
                                  (const float*)d_in[49], bnout, 64, 10, 0);
  lsm_k<<<1, 64, 0, stream>>>(bnout, out, regc);
}
